// crossattention1_66855460929913
// MI455X (gfx1250) — matrix operand order repaired and checked by reading, not compiled, not run on hardware
//
#include <hip/hip_runtime.h>
#include <cstdint>

// ---------------------------------------------------------------------------
// CDNA5 (gfx1250, wave32) cross-attention:
//   q = Q@Wq^T + bq ; k = K@Wk^T + bk ; v = K@Wv^T + bv
//   S = q k^T / sqrt(C) ; A = softmax(S) ; O = A v ; A_avg = head-major mean
// Strategy: bf16 WMMA (v_wmma_f32_16x16x32_bf16) everywhere, f32 accumulate.
// Projection GEMM operands are staged into LDS with the Tensor Data Mover
// (tensor_load_to_lds + s_wait_tensorcnt), double-buffered with the last
// k-step peeled so the hot loop has an unconditional s_wait_tensorcnt(2).
// q/k/v (~24MB bf16) are L2-resident (192MB), so the A_avg / O passes
// recompute score tiles instead of materializing the 256MB attention matrix.
// No atomics -> deterministic.
// ---------------------------------------------------------------------------

typedef __attribute__((ext_vector_type(16))) __bf16 v16bf;
typedef __attribute__((ext_vector_type(8)))  __bf16 bf16x8;
typedef __attribute__((ext_vector_type(8)))  float  v8f;
typedef unsigned int u32x4 __attribute__((ext_vector_type(4)));
typedef int          i32x4 __attribute__((ext_vector_type(4)));
typedef int          i32x8 __attribute__((ext_vector_type(8)));

#define DIM   1024
#define HEADS 16
#define BATCH 4
#define NP    1024
#define NR    1024
#define HD    64          // head dim
#define SCALE 0.03125f    // 1/sqrt(1024)

#if defined(__has_builtin)
#if __has_builtin(__builtin_amdgcn_tensor_load_to_lds) && \
    __has_builtin(__builtin_amdgcn_s_wait_tensorcnt)
#define HAS_TDM 1
#endif
#endif
#ifndef HAS_TDM
#define HAS_TDM 0
#endif

// ---- WMMA fragment loaders (per CDNA5 ISA 7.12.2 VGPR layouts) -------------
// A-matrix 16x32 bf16: lane<16 -> row M=lane, elems 0..7 = K kb..kb+7,
// elems 8..15 = K kb+16..kb+23, kb = (lane/16)*8.
__device__ __forceinline__ v16bf load_a_frag(const __bf16* rowp, int lane) {
  const int kb = (lane >> 4) * 8;
  bf16x8 lo = *(const bf16x8*)(rowp + kb);
  bf16x8 hi = *(const bf16x8*)(rowp + kb + 16);
  v16bf r;
#pragma unroll
  for (int i = 0; i < 8; ++i) { r[i] = lo[i]; r[8 + i] = hi[i]; }
  return r;
}
// B-matrix 32x16 bf16: col N=lane%16, elems 0..7 = K kb..kb+7,
// elems 8..15 = K kb+16..kb+23, kb = (lane/16)*8 (matches A-frag).
__device__ __forceinline__ v16bf load_b_frag(const __bf16* colp, int lane) {
  const int kb = (lane >> 4) * 8;
  bf16x8 lo = *(const bf16x8*)(colp + kb);
  bf16x8 hi = *(const bf16x8*)(colp + kb + 16);
  v16bf r;
#pragma unroll
  for (int i = 0; i < 8; ++i) { r[i] = lo[i]; r[8 + i] = hi[i]; }
  return r;
}

__device__ __forceinline__ v8f wmma_bf16(v16bf a, v16bf b, v8f c) {
  return __builtin_amdgcn_wmma_f32_16x16x32_bf16(false, a, false, b,
                                                 (short)0, c, false, false);
}

#if HAS_TDM
// ---- Tensor Data Mover: 2D tile load, bf16 elements ------------------------
// D# group0/group1 bit layout per CDNA5 ISA.
// Loads a tile_dim1 x tile_dim0 tile (rows x contiguous elems) whose first
// element is at `gaddr`, row stride `stride0` elements, packed row-major into
// LDS at byte offset `lds_off`.
__device__ __forceinline__ void tdm_load_2d(uint32_t lds_off, const void* gaddr,
                                            uint32_t tdim0, uint32_t tdim1,
                                            uint32_t stride0,
                                            uint32_t tile0, uint32_t tile1) {
  const uint64_t ga = (uint64_t)(uintptr_t)gaddr;
  u32x4 g0;
  g0[0] = 1u;                                        // count=1, user descriptor
  g0[1] = lds_off;                                   // lds_addr
  g0[2] = (uint32_t)(ga & 0xFFFFFFFFu);              // global_addr[31:0]
  g0[3] = (uint32_t)((ga >> 32) & 0x01FFFFFFu)       // global_addr[56:32]
          | (2u << 30);                              // type = 2 ("image")
  i32x8 g1;
  g1[0] = (int)(1u << 16);                           // data_size=1 -> 2 bytes
  g1[1] = (int)((tdim0 & 0xFFFFu) << 16);            // tensor_dim0[15:0]
  g1[2] = (int)((tdim0 >> 16) | ((tdim1 & 0xFFFFu) << 16));
  g1[3] = (int)((tdim1 >> 16) | (tile0 << 16));      // tile_dim0
  g1[4] = (int)(tile1 & 0xFFFFu);                    // tile_dim1 (tile_dim2=0)
  g1[5] = (int)stride0;                              // tensor_dim0_stride[31:0]
  g1[6] = 0;                                         // stride0[47:32], stride1 lo
  g1[7] = 0;                                         // stride1 hi
  i32x4 z4 = {0, 0, 0, 0};
#if defined(__clang_major__) && (__clang_major__ >= 23)
  i32x8 z8 = {0, 0, 0, 0, 0, 0, 0, 0};
  __builtin_amdgcn_tensor_load_to_lds(g0, g1, z4, z4, z8, 0);
#else
  __builtin_amdgcn_tensor_load_to_lds(g0, g1, z4, z4, 0);
#endif
}
#endif  // HAS_TDM

// ---- f32 -> bf16 conversion ------------------------------------------------
__global__ void k_cvt_bf16(const float* __restrict__ in, __bf16* __restrict__ out, int n) {
  int i = blockIdx.x * 256 + threadIdx.x;
  if (i < n) out[i] = (__bf16)in[i];
}

// ---- one k-step of the 64x64 projection tile: 8 A/B frags -> 16 WMMAs ------
__device__ __forceinline__ void proj_step(const __bf16* __restrict__ sAbuf,
                                          const __bf16* __restrict__ sBbuf,
                                          int rsel, int l, v8f c[4][4]) {
  v16bf a[4], b[4];
#pragma unroll
  for (int i = 0; i < 4; ++i)
    a[i] = load_a_frag(sAbuf + (i * 16 + rsel) * 32, l);
#pragma unroll
  for (int j = 0; j < 4; ++j)
    b[j] = load_b_frag(sBbuf + (j * 16 + rsel) * 32, l);
#pragma unroll
  for (int i = 0; i < 4; ++i)
#pragma unroll
    for (int j = 0; j < 4; ++j) c[i][j] = wmma_bf16(a[i], b[j], c[i][j]);
}

// ---- projection GEMM: Out[M=4096,N=1024] = A(bf16) x W^T + bias, bf16 out --
// One wave per 64x64 tile. K stepped by 32; A/B 64x32 tiles staged to LDS by
// the Tensor Data Mover, double buffered, consumed as WMMA fragments via ds.
// B-frag lanes read W rows directly (W is [n][c], exactly the K-contiguous
// per-column layout WMMA's B operand wants), so no transpose anywhere.
__global__ void __launch_bounds__(32)
k_proj(const __bf16* __restrict__ A, const __bf16* __restrict__ W,
       const float* __restrict__ bias, __bf16* __restrict__ Out) {
  const int l = threadIdx.x;
  const int rsel = l & 15, g = l >> 4;
  const int m0 = blockIdx.x * 64;
  const int n0 = blockIdx.y * 64;

  v8f c[4][4];
#pragma unroll
  for (int i = 0; i < 4; ++i)
#pragma unroll
    for (int j = 0; j < 4; ++j) c[i][j] = (v8f)0.0f;

#if HAS_TDM
  __shared__ __align__(16) __bf16 sA[2][64 * 32];   // 4KB x2
  __shared__ __align__(16) __bf16 sB[2][64 * 32];   // 4KB x2
  const uint32_t oA0 = (uint32_t)(uintptr_t)&sA[0][0];
  const uint32_t oA1 = (uint32_t)(uintptr_t)&sA[1][0];
  const uint32_t oB0 = (uint32_t)(uintptr_t)&sB[0][0];
  const uint32_t oB1 = (uint32_t)(uintptr_t)&sB[1][0];
  const __bf16* Abase = A + (size_t)m0 * DIM;
  const __bf16* Wbase = W + (size_t)n0 * DIM;

  // prime buffer 0 with the k=0 tiles
  tdm_load_2d(oA0, Abase, DIM, 4096, DIM, 32, 64);
  tdm_load_2d(oB0, Wbase, DIM, DIM,  DIM, 32, 64);

  // steady state: last k-step peeled -> unconditional wait(2) in the loop
  int buf = 0;
  for (int kk = 0; kk < DIM - 32; kk += 32, buf ^= 1) {
    tdm_load_2d(buf ? oA0 : oA1, Abase + kk + 32, DIM, 4096, DIM, 32, 64);
    tdm_load_2d(buf ? oB0 : oB1, Wbase + kk + 32, DIM, DIM,  DIM, 32, 64);
    __builtin_amdgcn_s_wait_tensorcnt(2);     // current buffer's 2 DMAs done
    asm volatile("" ::: "memory");            // order ds reads after the wait
    proj_step(&sA[buf][0], &sB[buf][0], rsel, l, c);
  }
  // epilogue: final k-step
  __builtin_amdgcn_s_wait_tensorcnt(0);
  asm volatile("" ::: "memory");
  proj_step(&sA[buf][0], &sB[buf][0], rsel, l, c);
#else
  for (int kk = 0; kk < DIM; kk += 32) {
    v16bf a[4], b[4];
#pragma unroll
    for (int i = 0; i < 4; ++i) {
      const __bf16* p = A + (size_t)(m0 + i * 16 + rsel) * DIM + kk;
      __builtin_prefetch(p + 32, 0, 1);
      a[i] = load_a_frag(p, l);
    }
#pragma unroll
    for (int j = 0; j < 4; ++j)
      b[j] = load_b_frag(W + (size_t)(n0 + j * 16 + rsel) * DIM + kk, l);
#pragma unroll
    for (int i = 0; i < 4; ++i)
#pragma unroll
      for (int j = 0; j < 4; ++j) c[i][j] = wmma_bf16(a[i], b[j], c[i][j]);
  }
#endif

#pragma unroll
  for (int i = 0; i < 4; ++i)
#pragma unroll
    for (int j = 0; j < 4; ++j) {
      const int ng = n0 + j * 16 + rsel;
      const float bv = bias[ng];
#pragma unroll
      for (int r = 0; r < 8; ++r) {
        const int mg = m0 + i * 16 + r + 8 * g;
        Out[(size_t)mg * DIM + ng] = (__bf16)(c[i][j][r] + bv);
      }
    }
}

// ---- vT[b,h,dd,n] = v[b,n,h*64+dd]  (B-operand layout for A@V) -------------
__global__ void k_transpose_v(const __bf16* __restrict__ v, __bf16* __restrict__ vT) {
  const int bh = blockIdx.x, b = bh >> 4, h = bh & 15;
  for (int i = threadIdx.x; i < HD * NR; i += 256) {
    const int dd = i >> 10, n = i & (NR - 1);
    vT[(size_t)(bh * HD + dd) * NR + n] =
        v[((size_t)(b * NR + n)) * DIM + h * HD + dd];
  }
}

// ---- pass 1: per-row softmax stats (max, sum) via WMMA score tiles ---------
__global__ void __launch_bounds__(32)
k_rowstats(const __bf16* __restrict__ q, const __bf16* __restrict__ kmat,
           float* __restrict__ Mrow, float* __restrict__ Lrow) {
  const int bh = blockIdx.x, b = bh >> 4, h = bh & 15;
  const int m0 = blockIdx.y * 16;
  const int l = threadIdx.x, rsel = l & 15, g = l >> 4;

  const __bf16* qrow = q + (size_t)(b * NP + m0 + rsel) * DIM + h * HD;
  const v16bf a0 = load_a_frag(qrow, l);        // d = 0..31
  const v16bf a1 = load_a_frag(qrow + 32, l);   // d = 32..63

  float mx[8], sm[8];
#pragma unroll
  for (int r = 0; r < 8; ++r) { mx[r] = -1e30f; sm[r] = 0.0f; }

  for (int nt = 0; nt < NR / 16; ++nt) {
    const __bf16* kcol = kmat + (size_t)(b * NR + nt * 16 + rsel) * DIM + h * HD;
    const v16bf b0 = load_b_frag(kcol, l);
    const v16bf b1 = load_b_frag(kcol + 32, l);
    v8f c = (v8f)0.0f;
    c = wmma_bf16(a0, b0, c);
    c = wmma_bf16(a1, b1, c);
#pragma unroll
    for (int r = 0; r < 8; ++r) {
      const float s = c[r] * SCALE;
      float tmax = s;
#pragma unroll
      for (int off = 8; off > 0; off >>= 1)
        tmax = fmaxf(tmax, __shfl_xor(tmax, off, 16));
      const float nm = fmaxf(mx[r], tmax);
      float ps = __expf(s - nm);
#pragma unroll
      for (int off = 8; off > 0; off >>= 1)
        ps += __shfl_xor(ps, off, 16);
      sm[r] = sm[r] * __expf(mx[r] - nm) + ps;
      mx[r] = nm;
    }
  }
  if (rsel == 0) {
#pragma unroll
    for (int r = 0; r < 8; ++r) {
      const int row = m0 + r + 8 * g;
      Mrow[(size_t)bh * NP + row] = mx[r];
      Lrow[(size_t)bh * NP + row] = sm[r];
    }
  }
}

// ---- pass 2a: A_avg[b_new] = mean over 16 (b,h) pairs (flat = h*B + b) -----
__global__ void __launch_bounds__(32)
k_Aavg(const __bf16* __restrict__ q, const __bf16* __restrict__ kmat,
       const float* __restrict__ Mrow, const float* __restrict__ Lrow,
       float* __restrict__ Aout) {
  __shared__ __align__(16) float tile[16][256];
  const int l = threadIdx.x, rsel = l & 15, g = l >> 4;
  const int bnew = blockIdx.x;
  const int m0 = blockIdx.y * 16;
  const int n0 = blockIdx.z * 256;

  for (int i = l; i < 16 * 256; i += 32) ((float*)tile)[i] = 0.0f;
  __syncthreads();

  for (int p = 0; p < 16; ++p) {
    const int flat = bnew * 16 + p;
    const int h = flat >> 2, b = flat & 3, bh = b * HEADS + h;
    const __bf16* qrow = q + (size_t)(b * NP + m0 + rsel) * DIM + h * HD;
    const v16bf a0 = load_a_frag(qrow, l);
    const v16bf a1 = load_a_frag(qrow + 32, l);
    float mi[8], li[8];
#pragma unroll
    for (int r = 0; r < 8; ++r) {
      const int row = m0 + r + 8 * g;
      mi[r] = Mrow[(size_t)bh * NP + row];
      li[r] = 1.0f / Lrow[(size_t)bh * NP + row];
    }
    for (int nt = 0; nt < 16; ++nt) {
      const int n = n0 + nt * 16;
      const __bf16* kcol = kmat + (size_t)(b * NR + n + rsel) * DIM + h * HD;
      v8f c = (v8f)0.0f;
      c = wmma_bf16(a0, load_b_frag(kcol, l), c);
      c = wmma_bf16(a1, load_b_frag(kcol + 32, l), c);
#pragma unroll
      for (int r = 0; r < 8; ++r) {
        const float aval = __expf(c[r] * SCALE - mi[r]) * li[r] * (1.0f / 16.0f);
        tile[r + 8 * g][nt * 16 + rsel] += aval;   // per-lane exclusive address
      }
    }
  }
  __syncthreads();
  float* dst = Aout + ((size_t)(bnew * NP + m0)) * NR + n0;
  for (int i = l; i < 16 * 256; i += 32) {
    const int rr = i >> 8, cc = i & 255;
    dst[(size_t)rr * NR + cc] = tile[rr][cc];
  }
}

// ---- pass 2b: O = A @ v  (A recomputed, routed via LDS to A-frag layout) ---
__global__ void __launch_bounds__(32)
k_Ov(const __bf16* __restrict__ q, const __bf16* __restrict__ kmat,
     const __bf16* __restrict__ vT,
     const float* __restrict__ Mrow, const float* __restrict__ Lrow,
     float* __restrict__ Oout) {
  __shared__ __align__(16) __bf16 atile[16][32];
  const int bh = blockIdx.x, b = bh >> 4, h = bh & 15;
  const int m0 = blockIdx.y * 16;
  const int l = threadIdx.x, rsel = l & 15, g = l >> 4;

  const __bf16* qrow = q + (size_t)(b * NP + m0 + rsel) * DIM + h * HD;
  const v16bf a0 = load_a_frag(qrow, l);
  const v16bf a1 = load_a_frag(qrow + 32, l);

  float mi[8], li[8];
#pragma unroll
  for (int r = 0; r < 8; ++r) {
    const int row = m0 + r + 8 * g;
    mi[r] = Mrow[(size_t)bh * NP + row];
    li[r] = 1.0f / Lrow[(size_t)bh * NP + row];
  }

  v8f acc[4];
#pragma unroll
  for (int j = 0; j < 4; ++j) acc[j] = (v8f)0.0f;

  for (int n = 0; n < NR; n += 32) {
#pragma unroll
    for (int half = 0; half < 2; ++half) {
      const int ncol = n + half * 16;
      const __bf16* kcol = kmat + (size_t)(b * NR + ncol + rsel) * DIM + h * HD;
      v8f c = (v8f)0.0f;
      c = wmma_bf16(a0, load_b_frag(kcol, l), c);
      c = wmma_bf16(a1, load_b_frag(kcol + 32, l), c);
#pragma unroll
      for (int r = 0; r < 8; ++r) {
        const float aval = __expf(c[r] * SCALE - mi[r]) * li[r];
        atile[r + 8 * g][half * 16 + rsel] = (__bf16)aval;  // C-layout -> LDS
      }
    }
    __syncthreads();
    const v16bf af = load_a_frag(&atile[rsel][0], l);       // LDS -> A-layout
    __syncthreads();
#pragma unroll
    for (int j = 0; j < 4; ++j) {
      const __bf16* vcol = vT + ((size_t)bh * HD + j * 16 + rsel) * NR + n;
      acc[j] = wmma_bf16(af, load_b_frag(vcol, l), acc[j]);
    }
  }
#pragma unroll
  for (int j = 0; j < 4; ++j)
#pragma unroll
    for (int r = 0; r < 8; ++r) {
      const int mg = m0 + r + 8 * g;
      const int dd = j * 16 + rsel;
      Oout[((size_t)(b * NP + mg)) * DIM + h * HD + dd] = acc[j][r];
    }
}

// ---------------------------------------------------------------------------
extern "C" void kernel_launch(void* const* d_in, const int* in_sizes, int n_in,
                              void* d_out, int out_size, void* d_ws, size_t ws_size,
                              hipStream_t stream) {
  const float* Q  = (const float*)d_in[0];   // [4,1024,1024]
  const float* K  = (const float*)d_in[1];   // [4,1024,1024]
  const float* Wq = (const float*)d_in[2];
  const float* bq = (const float*)d_in[3];
  const float* Wk = (const float*)d_in[4];
  const float* bk = (const float*)d_in[5];
  const float* Wv = (const float*)d_in[6];
  const float* bv = (const float*)d_in[7];

  char* ws = (char*)d_ws;
  const size_t SZ_QK = (size_t)BATCH * NP * DIM * sizeof(__bf16);   // 8 MB
  const size_t SZ_W  = (size_t)DIM * DIM * sizeof(__bf16);          // 2 MB
  const size_t SZ_ST = (size_t)BATCH * HEADS * NP * sizeof(float);  // 256 KB

  __bf16* Qb  = (__bf16*)(ws);                 ws += SZ_QK;
  __bf16* Kb  = (__bf16*)(ws);                 ws += SZ_QK;
  __bf16* Wqb = (__bf16*)(ws);                 ws += SZ_W;
  __bf16* Wkb = (__bf16*)(ws);                 ws += SZ_W;
  __bf16* Wvb = (__bf16*)(ws);                 ws += SZ_W;
  __bf16* qp  = (__bf16*)(ws);                 ws += SZ_QK;
  __bf16* kp  = (__bf16*)(ws);                 ws += SZ_QK;
  __bf16* vp  = (__bf16*)(ws);                 ws += SZ_QK;
  __bf16* vT  = (__bf16*)(ws);                 ws += SZ_QK;
  float*  Mrow = (float*)(ws);                 ws += SZ_ST;
  float*  Lrow = (float*)(ws);                 ws += SZ_ST;

  float* Oout = (float*)d_out;                               // [4,1024,1024]
  float* Aavg = (float*)d_out + (size_t)BATCH * NP * DIM;    // [4,1024,1024]

  // 1) f32 -> bf16
  const int nQK = BATCH * NP * DIM, nW = DIM * DIM;
  k_cvt_bf16<<<(nQK + 255) / 256, 256, 0, stream>>>(Q,  Qb,  nQK);
  k_cvt_bf16<<<(nQK + 255) / 256, 256, 0, stream>>>(K,  Kb,  nQK);
  k_cvt_bf16<<<(nW  + 255) / 256, 256, 0, stream>>>(Wq, Wqb, nW);
  k_cvt_bf16<<<(nW  + 255) / 256, 256, 0, stream>>>(Wk, Wkb, nW);
  k_cvt_bf16<<<(nW  + 255) / 256, 256, 0, stream>>>(Wv, Wvb, nW);

  // 2) projections (64x64 tile per wave, TDM-staged LDS operands): q, k, v
  dim3 gProj((BATCH * NP) / 64, DIM / 64);
  k_proj<<<gProj, 32, 0, stream>>>(Qb, Wqb, bq, qp);
  k_proj<<<gProj, 32, 0, stream>>>(Kb, Wkb, bk, kp);
  k_proj<<<gProj, 32, 0, stream>>>(Kb, Wvb, bv, vp);

  // 3) vT for A@V B-operand
  k_transpose_v<<<BATCH * HEADS, 256, 0, stream>>>(vp, vT);

  // 4) softmax row stats
  dim3 gStat(BATCH * HEADS, NP / 16);
  k_rowstats<<<gStat, 32, 0, stream>>>(qp, kp, Mrow, Lrow);

  // 5) A_avg (head-major mean, deterministic, no atomics)
  dim3 gAvg(BATCH, NP / 16, NR / 256);
  k_Aavg<<<gAvg, 32, 0, stream>>>(qp, kp, Mrow, Lrow, Aavg);

  // 6) O = A @ v
  dim3 gO(BATCH * HEADS, NP / 16);
  k_Ov<<<gO, 32, 0, stream>>>(qp, kp, vT, Mrow, Lrow, Oout);
}
